// TransformerEncoder_89146341196503
// MI455X (gfx1250) — matrix operand order repaired and checked by reading, not compiled, not run on hardware
//
#include <hip/hip_runtime.h>

typedef __attribute__((ext_vector_type(16))) __bf16 v16bf;
typedef __attribute__((ext_vector_type(8)))  float  v8f;
typedef __attribute__((ext_vector_type(4)))  unsigned int v4u;
typedef __attribute__((ext_vector_type(8)))  int   v8i;
typedef __attribute__((ext_vector_type(4)))  int   v4i;

union Frag32 {
  v16bf v;
  uint4 q[2];
  unsigned short h[16];
};

__device__ __forceinline__ unsigned short f32_to_bf16(float f) {
  unsigned int u = __float_as_uint(f);
  unsigned int r = (u + 0x7FFFu + ((u >> 16) & 1u)) >> 16;
  return (unsigned short)r;
}

constexpr int TT    = 2048;
constexpr int EE    = 256;
constexpr int HH    = 8;
constexpr int DHEAD = 32;
constexpr int FFN   = 1024;
constexpr int NLAYER = 2;
constexpr int QKVW  = 3 * EE;   // fused q|k|v projection width = 768

#if __has_builtin(__builtin_amdgcn_tensor_load_to_lds)
#define USE_TDM 1
#else
#define USE_TDM 0
#endif

#if USE_TDM
// ---------------------------------------------------------------------------
// Issue one TDM load of a [128 rows x 32 elem] bf16 tile (row stride = K
// elements) from global memory into LDS at byte offset ldsAddr. 2D tile:
// groups 2/3 unused (zero). Tracked by TENSORcnt.
// D# packing per CDNA5 ISA.
// ---------------------------------------------------------------------------
__device__ __forceinline__ void tdm_load_tile(const unsigned short* g,
                                              unsigned ldsAddr, int K) {
  unsigned long long ga = (unsigned long long)g;
  v4u g0;
  g0.x = 1u;                                   // count=1, is_restore=0
  g0.y = ldsAddr;                              // lds_addr (bytes)
  g0.z = (unsigned)ga;                         // global_addr[31:0]
  g0.w = (unsigned)(ga >> 32) | 0x80000000u;   // global_addr[56:32] | type=2
  v8i g1;
  g1[0] = 1 << 16;                             // data_size = 2 bytes
  g1[1] = (K & 0xFFFF) << 16;                  // tensor_dim0[15:0] = K
  g1[2] = ((K >> 16) & 0xFFFF) | (0xFFFF << 16); // td0 hi | tensor_dim1 lo
  g1[3] = (32 << 16);                          // td1 hi=0 | tile_dim0 = 32
  g1[4] = 128;                                 // tile_dim1 = 128, tile_dim2=0
  g1[5] = K;                                   // tensor_dim0_stride[31:0]
  g1[6] = 0;                                   // stride0 hi | stride1 lo
  g1[7] = 0;
  v4i z4 = {};
#if __clang_major__ >= 23
  v8i z8 = {};
  __builtin_amdgcn_tensor_load_to_lds(g0, g1, z4, z4, z8, 0);
#else
  __builtin_amdgcn_tensor_load_to_lds(g0, g1, z4, z4, 0);
#endif
}
#endif

// ---------------------------------------------------------------------------
// Embedding + sinusoid positional encoding: h = (x @ w_emb)*sqrt(E) + pe
// ---------------------------------------------------------------------------
__global__ __launch_bounds__(256)
void embed_pe_kernel(const float* __restrict__ x,
                     const float* __restrict__ wemb,
                     float* __restrict__ hf,
                     unsigned short* __restrict__ hb) {
  int t = blockIdx.x;
  int e = threadIdx.x;
  const float* xr = x + (size_t)t * 64;
  float acc = 0.f;
#pragma unroll 8
  for (int i = 0; i < 64; ++i) acc = fmaf(xr[i], wemb[(size_t)i * EE + e], acc);
  acc *= 16.0f;  // sqrt(256)
  float pos = (float)(TT - 1 - t);   // causal: descending positions
  int j = e & 127;
  // inv_freq = 10000^(-2j/256)
  float invf = __expf(-((float)(2 * j) / (float)EE) * 9.210340371976184f);
  float ang = pos * invf;
  float pe = (e < 128) ? __sinf(ang) : __cosf(ang);
  float v = acc + pe;
  size_t o = (size_t)t * EE + e;
  hf[o] = v;
  hb[o] = f32_to_bf16(v);
}

// ---------------------------------------------------------------------------
// fp32 [K,N] weight -> bf16 transposed [N,K] (one-time per launch)
// ---------------------------------------------------------------------------
__global__ void convert_transpose_kernel(const float* __restrict__ W,
                                         unsigned short* __restrict__ Wt,
                                         int K, int N) {
  int idx = blockIdx.x * blockDim.x + threadIdx.x;
  if (idx >= K * N) return;
  int k = idx / N, n = idx - k * N;
  Wt[(size_t)n * K + k] = f32_to_bf16(W[idx]);
}

// ---------------------------------------------------------------------------
// WMMA GEMM: C[M,N] = A[M,K](bf16) @ Bt[N,K](bf16)^T, optional bias/relu,
// fp32 and/or bf16 outputs. Block = 256 thr (8 waves), tile 128x128, BK=32.
// Wave grid 2x4 -> each wave 64x32 = 4x2 WMMA 16x16 tiles.
// Tiles staged into double-buffered LDS by the Tensor Data Mover (wave 0
// issues tensor_load_to_lds; TENSORcnt + barrier publish). DMA of tile i+1
// overlaps WMMA on tile i.
// ---------------------------------------------------------------------------
__global__ __launch_bounds__(256)
void wmma_gemm_kernel(const unsigned short* __restrict__ A,
                      const unsigned short* __restrict__ Bt,
                      const float* __restrict__ bias,
                      float* __restrict__ Cf,
                      unsigned short* __restrict__ Cb,
                      int M, int N, int K, int relu) {
  __shared__ unsigned short As[2][128 * 32];   // [buf][row][k] m-major
  __shared__ unsigned short Bs[2][128 * 32];   // [buf][col][k] n-major
  int tid  = threadIdx.x;
  int wave = tid >> 5, lane = tid & 31;
  int wm = wave >> 2, wn = wave & 3;
  int g = lane >> 4, ln = lane & 15;
  int bm = blockIdx.y * 128, bn = blockIdx.x * 128;

  v8f acc[4][2] = {};

  const unsigned short* Ag = A  + (size_t)bm * K;   // block A rows
  const unsigned short* Bg = Bt + (size_t)bn * K;   // block B cols
  const int nit = K >> 5;

#if USE_TDM
  unsigned asBase = (unsigned)(unsigned long long)(const void*)&As[0][0];
  unsigned bsBase = (unsigned)(unsigned long long)(const void*)&Bs[0][0];
  // prologue: DMA tile 0 into buffer 0
  if (wave == 0) {
    tdm_load_tile(Ag, asBase, K);
    tdm_load_tile(Bg, bsBase, K);
    __builtin_amdgcn_s_wait_tensorcnt(0);
  }
  __syncthreads();
#else
  {
    int lr = tid >> 1, lc = (tid & 1) * 16;
    *(uint4*)(&As[0][lr * 32 + lc])     = *(const uint4*)(Ag + (size_t)lr * K + lc);
    *(uint4*)(&As[0][lr * 32 + lc + 8]) = *(const uint4*)(Ag + (size_t)lr * K + lc + 8);
    *(uint4*)(&Bs[0][lr * 32 + lc])     = *(const uint4*)(Bg + (size_t)lr * K + lc);
    *(uint4*)(&Bs[0][lr * 32 + lc + 8]) = *(const uint4*)(Bg + (size_t)lr * K + lc + 8);
  }
  __syncthreads();
#endif

  for (int it = 0; it < nit; ++it) {
    int buf = it & 1;
    // kick off DMA for the next tile into the other buffer (overlaps compute)
    if (it + 1 < nit) {
#if USE_TDM
      if (wave == 0) {
        tdm_load_tile(Ag + (size_t)(it + 1) * 32, asBase + (unsigned)((it + 1) & 1) * 8192u, K);
        tdm_load_tile(Bg + (size_t)(it + 1) * 32, bsBase + (unsigned)((it + 1) & 1) * 8192u, K);
      }
#else
      int nb = (it + 1) & 1, k0 = (it + 1) * 32;
      int lr = tid >> 1, lc = (tid & 1) * 16;
      *(uint4*)(&As[nb][lr * 32 + lc])     = *(const uint4*)(Ag + (size_t)lr * K + k0 + lc);
      *(uint4*)(&As[nb][lr * 32 + lc + 8]) = *(const uint4*)(Ag + (size_t)lr * K + k0 + lc + 8);
      *(uint4*)(&Bs[nb][lr * 32 + lc])     = *(const uint4*)(Bg + (size_t)lr * K + k0 + lc);
      *(uint4*)(&Bs[nb][lr * 32 + lc + 8]) = *(const uint4*)(Bg + (size_t)lr * K + k0 + lc + 8);
#endif
    }

    const unsigned short* Asb = &As[buf][0];
    const unsigned short* Bsb = &Bs[buf][0];

    // B fragments: lane = col n, element i -> k = 16*g + i (contiguous)
    Frag32 bf0, bf1;
    {
      const unsigned short* r0 = Bsb + (wn * 32 + ln) * 32 + g * 16;
      bf0.q[0] = *(const uint4*)(r0);
      bf0.q[1] = *(const uint4*)(r0 + 8);
      const unsigned short* r1 = Bsb + (wn * 32 + 16 + ln) * 32 + g * 16;
      bf1.q[0] = *(const uint4*)(r1);
      bf1.q[1] = *(const uint4*)(r1 + 8);
    }
#pragma unroll
    for (int ms = 0; ms < 4; ++ms) {
      // A fragment: lane = row m, element i -> k = 16*g + i (contiguous)
      Frag32 af;
      const unsigned short* row = Asb + (wm * 64 + ms * 16 + ln) * 32;
      af.q[0] = *(const uint4*)(row + g * 16);
      af.q[1] = *(const uint4*)(row + g * 16 + 8);
      acc[ms][0] = __builtin_amdgcn_wmma_f32_16x16x32_bf16(
          false, af.v, false, bf0.v, (short)0, acc[ms][0], false, false);
      acc[ms][1] = __builtin_amdgcn_wmma_f32_16x16x32_bf16(
          false, af.v, false, bf1.v, (short)0, acc[ms][1], false, false);
    }

#if USE_TDM
    if (wave == 0) __builtin_amdgcn_s_wait_tensorcnt(0);
#endif
    __syncthreads();   // publish next tile / protect buffers
  }

#pragma unroll
  for (int ms = 0; ms < 4; ++ms) {
#pragma unroll
    for (int ns = 0; ns < 2; ++ns) {
      int c = bn + wn * 32 + ns * 16 + ln;
      float bv = bias ? bias[c] : 0.f;
#pragma unroll
      for (int j = 0; j < 8; ++j) {
        int r = bm + wm * 64 + ms * 16 + j + 8 * g;  // C layout: M = j + 8g
        float v = acc[ms][ns][j] + bv;
        if (relu) v = fmaxf(v, 0.f);
        size_t o = (size_t)r * N + c;
        if (Cf) Cf[o] = v;
        if (Cb) Cb[o] = f32_to_bf16(v);
      }
    }
  }
}

// ---------------------------------------------------------------------------
// Banded flash attention: one wave per (head, 16-query tile).
// Band +-128 -> 272 keys; processed as 9 pairs of 16-key tiles starting
// at qbase-128 (aligned), invalid keys masked. Scores & P@V via WMMA.
// qkv is the fused [T, 768] bf16 projection (Q | K | V columns).
// ---------------------------------------------------------------------------
__global__ __launch_bounds__(32)
void wmma_attn_kernel(const unsigned short* __restrict__ qkv,
                      unsigned short* __restrict__ attn) {
  __shared__ unsigned short Pl[16 * 32];   // P tile, row-major [query][key]
  int qt = blockIdx.x, hh = blockIdx.y;
  int lane = threadIdx.x;
  int g = lane >> 4, ln = lane & 15;
  int qbase = qt * 16;
  const float scale = 0.17677669529663687f;  // 1/sqrt(32)
  const unsigned short* Qb = qkv;
  const unsigned short* Kb = qkv + EE;
  const unsigned short* Vb = qkv + 2 * EE;

  // Q fragment (A layout), loaded once
  Frag32 qf;
  {
    const unsigned short* row = Qb + (size_t)(qbase + ln) * QKVW + hh * DHEAD;
    qf.q[0] = *(const uint4*)(row + g * 16);
    qf.q[1] = *(const uint4*)(row + g * 16 + 8);
  }

  v8f O0 = {}, O1 = {};
  float rm[8], rs[8];
#pragma unroll
  for (int j = 0; j < 8; ++j) { rm[j] = -1e30f; rs[j] = 0.f; }

  int kstart = qbase - 128;
  for (int pair = 0; pair < 9; ++pair) {
    int kb = kstart + pair * 32;
    v8f s0, s1;
    {
      int key = kb + ln;
      int kc = key < 0 ? 0 : (key > TT - 1 ? TT - 1 : key);
      const unsigned short* kr = Kb + (size_t)kc * QKVW + hh * DHEAD + g * 16;
      Frag32 bf; bf.q[0] = *(const uint4*)kr; bf.q[1] = *(const uint4*)(kr + 8);
      v8f z = {};
      s0 = __builtin_amdgcn_wmma_f32_16x16x32_bf16(
          false, qf.v, false, bf.v, (short)0, z, false, false);
    }
    {
      int key = kb + 16 + ln;
      int kc = key < 0 ? 0 : (key > TT - 1 ? TT - 1 : key);
      const unsigned short* kr = Kb + (size_t)kc * QKVW + hh * DHEAD + g * 16;
      Frag32 bf; bf.q[0] = *(const uint4*)kr; bf.q[1] = *(const uint4*)(kr + 8);
      v8f z = {};
      s1 = __builtin_amdgcn_wmma_f32_16x16x32_bf16(
          false, qf.v, false, bf.v, (short)0, z, false, false);
    }

    // online softmax over this pair of key tiles (rows j+8g, cols across lanes)
#pragma unroll
    for (int j = 0; j < 8; ++j) {
      int t = qbase + j + 8 * g;
      int k0i = kb + ln, k1i = kb + 16 + ln;
      int d0 = k0i - t; if (d0 < 0) d0 = -d0;
      int d1 = k1i - t; if (d1 < 0) d1 = -d1;
      bool v0 = (k0i >= 0) && (k0i < TT) && (d0 <= 128);
      bool v1 = (k1i >= 0) && (k1i < TT) && (d1 <= 128);
      float a0 = v0 ? s0[j] * scale : -1e30f;
      float a1 = v1 ? s1[j] * scale : -1e30f;
      float m = fmaxf(a0, a1);
      m = fmaxf(m, __shfl_xor(m, 1, 32));
      m = fmaxf(m, __shfl_xor(m, 2, 32));
      m = fmaxf(m, __shfl_xor(m, 4, 32));
      m = fmaxf(m, __shfl_xor(m, 8, 32));
      float nm = fmaxf(rm[j], m);
      float alpha = __expf(rm[j] - nm);
      float p0 = v0 ? __expf(a0 - nm) : 0.f;
      float p1 = v1 ? __expf(a1 - nm) : 0.f;
      float ls = p0 + p1;
      ls += __shfl_xor(ls, 1, 32);
      ls += __shfl_xor(ls, 2, 32);
      ls += __shfl_xor(ls, 4, 32);
      ls += __shfl_xor(ls, 8, 32);
      rs[j] = rs[j] * alpha + ls;
      rm[j] = nm;
      O0[j] *= alpha;
      O1[j] *= alpha;
      Pl[(j + 8 * g) * 32 + ln]      = f32_to_bf16(p0);
      Pl[(j + 8 * g) * 32 + 16 + ln] = f32_to_bf16(p1);
    }
    __syncthreads();

    // P as A fragment (16 queries x 32 keys)
    Frag32 pf;
    {
      const unsigned short* row = Pl + ln * 32;
      pf.q[0] = *(const uint4*)(row + g * 16);
      pf.q[1] = *(const uint4*)(row + g * 16 + 8);
    }
    // V as two B fragments (d = 0..15 and 16..31), contraction k = 16g+i
    Frag32 vf0, vf1;
#pragma unroll
    for (int i = 0; i < 16; ++i) {
      int key = kb + g * 16 + i;
      int kc = key < 0 ? 0 : (key > TT - 1 ? TT - 1 : key);
      const unsigned short* vr = Vb + (size_t)kc * QKVW + hh * DHEAD;
      vf0.h[i] = vr[ln];
      vf1.h[i] = vr[16 + ln];
    }
    O0 = __builtin_amdgcn_wmma_f32_16x16x32_bf16(
        false, pf.v, false, vf0.v, (short)0, O0, false, false);
    O1 = __builtin_amdgcn_wmma_f32_16x16x32_bf16(
        false, pf.v, false, vf1.v, (short)0, O1, false, false);
    __syncthreads();
  }

#pragma unroll
  for (int j = 0; j < 8; ++j) {
    float inv = 1.0f / rs[j];
    int t = qbase + j + 8 * g;
    unsigned short* orow = attn + (size_t)t * EE + hh * DHEAD;
    orow[ln]      = f32_to_bf16(O0[j] * inv);
    orow[16 + ln] = f32_to_bf16(O1[j] * inv);
  }
}

// ---------------------------------------------------------------------------
// out = LayerNorm(a + b) * scale + offset  (one block per row, E=256)
// ---------------------------------------------------------------------------
__global__ __launch_bounds__(256)
void add_layernorm_kernel(const float* __restrict__ a, const float* __restrict__ b,
                          const float* __restrict__ sc, const float* __restrict__ of,
                          float* __restrict__ outf, unsigned short* __restrict__ outb) {
  __shared__ float red[256];
  int t = blockIdx.x, e = threadIdx.x;
  size_t o = (size_t)t * EE + e;
  float v = a[o] + b[o];
  red[e] = v;
  __syncthreads();
  for (int st = 128; st > 0; st >>= 1) {
    if (e < st) red[e] += red[e + st];
    __syncthreads();
  }
  float mean = red[0] * (1.0f / EE);
  __syncthreads();
  float d = v - mean;
  red[e] = d * d;
  __syncthreads();
  for (int st = 128; st > 0; st >>= 1) {
    if (e < st) red[e] += red[e + st];
    __syncthreads();
  }
  float var = red[0] * (1.0f / EE);
  float r = rsqrtf(var + 1e-5f);
  float out = d * r * sc[e] + of[e];
  outf[o] = out;
  outb[o] = f32_to_bf16(out);
}

// ---------------------------------------------------------------------------
extern "C" void kernel_launch(void* const* d_in, const int* in_sizes, int n_in,
                              void* d_out, int out_size, void* d_ws, size_t ws_size,
                              hipStream_t stream) {
  (void)in_sizes; (void)n_in; (void)out_size; (void)ws_size;
  const float* x    = (const float*)d_in[0];
  const float* wemb = (const float*)d_in[1];
  const float* wq   = (const float*)d_in[2];
  const float* wk   = (const float*)d_in[3];
  const float* wv   = (const float*)d_in[4];
  const float* wo   = (const float*)d_in[5];
  const float* w1   = (const float*)d_in[6];
  const float* b1   = (const float*)d_in[7];
  const float* w2   = (const float*)d_in[8];
  const float* b2   = (const float*)d_in[9];
  const float* ln1s = (const float*)d_in[10];
  const float* ln1o = (const float*)d_in[11];
  const float* ln2s = (const float*)d_in[12];
  const float* ln2o = (const float*)d_in[13];
  float* out = (float*)d_out;

  char* ws = (char*)d_ws;
  size_t off = 0;
  auto alloc = [&](size_t bytes) -> void* {
    void* p = ws + off;
    off += (bytes + 255) & ~(size_t)255;
    return p;
  };
  float* hf = (float*)alloc((size_t)TT * EE * 4);
  float* af = (float*)alloc((size_t)TT * EE * 4);
  float* tf = (float*)alloc((size_t)TT * EE * 4);
  unsigned short* hb    = (unsigned short*)alloc((size_t)TT * EE * 2);
  unsigned short* ab    = (unsigned short*)alloc((size_t)TT * EE * 2);
  unsigned short* qkvb  = (unsigned short*)alloc((size_t)TT * QKVW * 2);
  unsigned short* attnb = (unsigned short*)alloc((size_t)TT * EE * 2);
  unsigned short* f1b   = (unsigned short*)alloc((size_t)TT * FFN * 2);
  unsigned short* wqkvT = (unsigned short*)alloc((size_t)NLAYER * QKVW * EE * 2);
  unsigned short* woT   = (unsigned short*)alloc((size_t)NLAYER * EE * EE * 2);
  unsigned short* w1T   = (unsigned short*)alloc((size_t)NLAYER * EE * FFN * 2);
  unsigned short* w2T   = (unsigned short*)alloc((size_t)NLAYER * FFN * EE * 2);

  // weight conversion: bf16, transposed to [N][K]; q|k|v fused per layer
  for (int l = 0; l < NLAYER; ++l) {
    size_t wof = (size_t)l * EE * EE;
    convert_transpose_kernel<<<(EE * EE + 255) / 256, 256, 0, stream>>>(
        wq + wof, wqkvT + (size_t)l * QKVW * EE, EE, EE);
    convert_transpose_kernel<<<(EE * EE + 255) / 256, 256, 0, stream>>>(
        wk + wof, wqkvT + (size_t)l * QKVW * EE + (size_t)EE * EE, EE, EE);
    convert_transpose_kernel<<<(EE * EE + 255) / 256, 256, 0, stream>>>(
        wv + wof, wqkvT + (size_t)l * QKVW * EE + (size_t)2 * EE * EE, EE, EE);
    convert_transpose_kernel<<<(EE * EE + 255) / 256, 256, 0, stream>>>(
        wo + wof, woT + wof, EE, EE);
    convert_transpose_kernel<<<(EE * FFN + 255) / 256, 256, 0, stream>>>(
        w1 + (size_t)l * EE * FFN, w1T + (size_t)l * EE * FFN, EE, FFN);
    convert_transpose_kernel<<<(FFN * EE + 255) / 256, 256, 0, stream>>>(
        w2 + (size_t)l * FFN * EE, w2T + (size_t)l * FFN * EE, FFN, EE);
  }

  embed_pe_kernel<<<TT, 256, 0, stream>>>(x, wemb, hf, hb);

  for (int l = 0; l < NLAYER; ++l) {
    // fused QKV projection: [T,256] x [256,768]
    wmma_gemm_kernel<<<dim3(QKVW / 128, TT / 128), 256, 0, stream>>>(
        hb, wqkvT + (size_t)l * QKVW * EE, nullptr, nullptr, qkvb,
        TT, QKVW, EE, 0);
    // banded attention
    wmma_attn_kernel<<<dim3(TT / 16, HH), 32, 0, stream>>>(qkvb, attnb);
    // output projection
    wmma_gemm_kernel<<<dim3(EE / 128, TT / 128), 256, 0, stream>>>(
        attnb, woT + (size_t)l * EE * EE, nullptr, tf, nullptr,
        TT, EE, EE, 0);
    // a = LN(h + attn_out)
    add_layernorm_kernel<<<TT, 256, 0, stream>>>(
        hf, tf, ln1s + (size_t)l * EE, ln1o + (size_t)l * EE, af, ab);
    // FFN up + ReLU (fused)
    wmma_gemm_kernel<<<dim3(FFN / 128, TT / 128), 256, 0, stream>>>(
        ab, w1T + (size_t)l * EE * FFN, b1 + (size_t)l * FFN, nullptr, f1b,
        TT, FFN, EE, 1);
    // FFN down
    wmma_gemm_kernel<<<dim3(EE / 128, TT / 128), 256, 0, stream>>>(
        f1b, w2T + (size_t)l * FFN * EE, b2 + (size_t)l * EE, tf, nullptr,
        TT, EE, FFN, 0);
    // h = LN(f + a); final layer writes straight to d_out
    float* outdst = (l == NLAYER - 1) ? out : hf;
    add_layernorm_kernel<<<TT, 256, 0, stream>>>(
        tf, af, ln2s + (size_t)l * EE, ln2o + (size_t)l * EE, outdst, hb);
  }
}
